// DHGLayerV1_39084202394052
// MI455X (gfx1250) — matrix operand order repaired and checked by reading, not compiled, not run on hardware
//
#include <hip/hip_runtime.h>
#include <hip/hip_bf16.h>
#include <math.h>

// DHGLayer forward for MI455X (gfx1250, wave32).
//   feats: [100000,128] f32; edge_members: [20000,8] i32; adj_members: [20000,4,8] i32
//   out: [20000,2] f32
// Workspace layout: ft = f32[20000*5*128] (51.2MB), sc = f32[100000].

typedef __attribute__((ext_vector_type(16))) _Float16 v16h;
typedef __attribute__((ext_vector_type(8)))  float    v8f;

#define DD   128
#define KK   8
#define RR   4
#define CAND 5          // 1 + R candidates per edge
#define WAVES_PER_BLOCK 8

static __device__ __forceinline__ float wave_allreduce(float v) {
    // full 32-lane butterfly; every lane ends with the total
    #pragma unroll
    for (int m = 16; m > 0; m >>= 1) v += __shfl_xor(v, m, 32);
    return v;
}

// ---------------------------------------------------------------------------
// Kernel 1: vertex_conv for all 100000 groups (one wave per group).
// Lane L owns columns {L, L+32, L+64, L+96}.
// ---------------------------------------------------------------------------
__global__ __launch_bounds__(256) void vconv_kernel(
    const float* __restrict__ feats,
    const int*   __restrict__ edge_members,   // [E*8]
    const int*   __restrict__ adj_members,    // [E*4*8]
    const float* __restrict__ wq, const float* __restrict__ bq,
    const float* __restrict__ wk, const float* __restrict__ bk,
    const float* __restrict__ wv, const float* __restrict__ bv,
    float* __restrict__ ft,                   // [E*5*128]
    int E)
{
    const int lane = threadIdx.x & 31;
    const int wave = threadIdx.x >> 5;
    const long g   = (long)blockIdx.x * WAVES_PER_BLOCK + wave;
    const long ngroups = (long)E * CAND;      // E + E*R
    if (g >= ngroups) return;

    const int* mem;
    float* outp;
    if (g < E) {                              // hyperedge itself -> slot 0
        mem  = edge_members + g * KK;
        outp = ft + g * (CAND * DD);
    } else {                                  // adjacent edge (e, r) -> slot 1+r
        long t = g - E;                       // t = e*R + r  (matches reshape(n*R,K))
        long e = t >> 2, r = t & 3;
        mem  = adj_members + t * KK;
        outp = ft + (e * CAND + 1 + r) * DD;
    }

    // gather 8 vertex rows, 4 columns per lane
    float f[KK][4];
    int   idx[KK];
    #pragma unroll
    for (int j = 0; j < KK; ++j) idx[j] = mem[j];
    #pragma unroll
    for (int j = 0; j < KK; ++j) {
        const float* row = feats + (long)idx[j] * DD;
        #pragma unroll
        for (int cc = 0; cc < 4; ++cc) f[j][cc] = row[cc * 32 + lane];
    }

    float wqv[4], wkv[4], wvv[4];
    #pragma unroll
    for (int cc = 0; cc < 4; ++cc) {
        int c = cc * 32 + lane;
        wqv[cc] = wq[c]; wkv[cc] = wk[c]; wvv[cc] = wv[c];
    }
    const float bqv = bq[0], bkv = bk[0], bvv = bv[0];

    // scalar q,k,v per vertex (wave-wide dot products)
    float q[KK], k[KK], v[KK];
    #pragma unroll
    for (int j = 0; j < KK; ++j) {
        float sq = 0.f, sk = 0.f, sv = 0.f;
        #pragma unroll
        for (int cc = 0; cc < 4; ++cc) {
            sq = fmaf(f[j][cc], wqv[cc], sq);
            sk = fmaf(f[j][cc], wkv[cc], sk);
            sv = fmaf(f[j][cc], wvv[cc], sv);
        }
        q[j] = wave_allreduce(sq) + bqv;
        k[j] = wave_allreduce(sk) + bkv;
        v[j] = wave_allreduce(sv) + bvv;
    }

    // masked 8x8 softmax + a@v + tanh gate (replicated across lanes; cheap)
    float di[KK];
    #pragma unroll
    for (int j = 0; j < KK; ++j) {
        float mx = -3.402823e38f;
        #pragma unroll
        for (int l = 0; l < KK; ++l)
            if (l != j) mx = fmaxf(mx, q[j] * k[l]);
        float den = 0.f, num = 0.f;
        #pragma unroll
        for (int l = 0; l < KK; ++l)
            if (l != j) {
                float e = __expf(q[j] * k[l] - mx);
                den += e; num = fmaf(e, v[l], num);
            }
        di[j] = tanhf(num / den);
    }

    #pragma unroll
    for (int cc = 0; cc < 4; ++cc) {
        float o = 0.f;
        #pragma unroll
        for (int j = 0; j < KK; ++j) o = fmaf(di[j], f[j][cc], o);
        outp[cc * 32 + lane] = o;
    }
}

// ---------------------------------------------------------------------------
// Kernel 2: EdgeConv scores via WMMA.
//   sc[m] = relu(ft[m,:] @ W1 + b1) @ W2 + b2  for m in [0, 100000)
// One wave per 16-row M tile; N=32 as two 16x16 C tiles; K=128 as 4 chunks
// of 32 with v_wmma_f32_16x16x32_f16 (f16 in, f32 accumulate).
// Launched with few waves (~784) so each wave strides over ~8 tiles,
// amortizing the tile-invariant W1 fragment setup; the next tile's A rows
// are prefetched (global_prefetch_b8) behind the current tile's WMMAs.
// ---------------------------------------------------------------------------
__global__ __launch_bounds__(256) void edgeconv_kernel(
    const float* __restrict__ ft,   // [nrows,128] (nrows = E*5)
    const float* __restrict__ W1,   // [128,32]
    const float* __restrict__ b1,   // [32]
    const float* __restrict__ W2,   // [32]
    const float* __restrict__ b2,   // [1]
    float* __restrict__ sc,         // [nrows]
    int nrows)
{
    const int lane  = threadIdx.x & 31;
    const int wave  = threadIdx.x >> 5;
    const int half  = lane >> 4;          // lanes 0-15 vs 16-31
    const int l16   = lane & 15;
    const int wid   = blockIdx.x * WAVES_PER_BLOCK + wave;
    const int nwv   = gridDim.x * WAVES_PER_BLOCK;

    // B fragments (W1) are tile-invariant: load once per wave.
    // B layout (16-bit, 32x16): t<8 -> K = half*8 + t, t>=8 -> K = 16 + half*8 + (t-8), lane's N column = l16 (+16 for tile 1).
    v16h bfrag[4][2];
    #pragma unroll
    for (int kc = 0; kc < 4; ++kc) {
        const int kbase = kc * 32 + half * 8;
        #pragma unroll
        for (int nt = 0; nt < 2; ++nt) {
            const int n = nt * 16 + l16;
            #pragma unroll
            for (int t = 0; t < 16; ++t)
                bfrag[kc][nt][t] = (_Float16)W1[(kbase + (t < 8 ? t : t + 8)) * 32 + n];
        }
    }
    const float b1A = b1[l16],      b1B = b1[16 + l16];
    const float w2A = W2[l16],      w2B = W2[16 + l16];
    const float b2v = b2[0];

    const int ntiles = nrows >> 4;                  // 6250
    for (int tile = wid; tile < ntiles; tile += nwv) {
        const float* base = ft + (long)tile * 16 * DD;

        // prefetch this wave's NEXT tile (L2-resident -> pull toward WGP)
        if (tile + nwv < ntiles) {
            const float* nbase = ft + (long)(tile + nwv) * 16 * DD;
            __builtin_prefetch(nbase + (long)l16 * DD + half * 64, 0, 3);
        }

        v8f acc0 = {}; v8f acc1 = {};
        #pragma unroll
        for (int kc = 0; kc < 4; ++kc) {
            // A layout (16-bit, 16x32): lane's row M = l16;
            // a[0..7] = K = kc*32 + half*8 + t ; a[8..15] = same + 16.
            v16h a;
            const float* rowp = base + (long)l16 * DD + kc * 32 + half * 8;
            #pragma unroll
            for (int t = 0; t < 8; ++t) {
                a[t]     = (_Float16)rowp[t];
                a[8 + t] = (_Float16)rowp[16 + t];
            }
            acc0 = __builtin_amdgcn_wmma_f32_16x16x32_f16(
                       false, a, false, bfrag[kc][0], (short)0, acc0, false, false);
            acc1 = __builtin_amdgcn_wmma_f32_16x16x32_f16(
                       false, a, false, bfrag[kc][1], (short)0, acc1, false, false);
        }
        // C layout: acc[r] holds M = r + 8*half, N = l16 (acc0) / 16+l16 (acc1).
        // relu + b1, project by W2, reduce over the 16 N-lanes in each half.
        float part[8];
        #pragma unroll
        for (int r = 0; r < 8; ++r) {
            float h0 = acc0[r] + b1A; h0 = h0 > 0.f ? h0 : 0.f;
            float h1 = acc1[r] + b1B; h1 = h1 > 0.f ? h1 : 0.f;
            part[r] = fmaf(h0, w2A, h1 * w2B);
        }
        #pragma unroll
        for (int mN = 1; mN < 16; mN <<= 1) {
            #pragma unroll
            for (int r = 0; r < 8; ++r) part[r] += __shfl_xor(part[r], mN, 32);
        }
        if (l16 == 0) {
            #pragma unroll
            for (int r = 0; r < 8; ++r)
                sc[tile * 16 + half * 8 + r] = part[r] + b2v;
        }
    }
}

// ---------------------------------------------------------------------------
// Kernel 3: softmax over the 5 candidates, weighted sum, sigmoid classifier.
// One wave per edge.
// ---------------------------------------------------------------------------
__global__ __launch_bounds__(256) void final_kernel(
    const float* __restrict__ ft,    // [E,5,128]
    const float* __restrict__ sc,    // [E,5]
    const float* __restrict__ Wfc,   // [128,2]
    const float* __restrict__ bfc,   // [2]
    float* __restrict__ out,         // [E,2]
    int E)
{
    const int lane = threadIdx.x & 31;
    const int wave = threadIdx.x >> 5;
    const long e   = (long)blockIdx.x * WAVES_PER_BLOCK + wave;
    if (e >= E) return;

    const float* scp = sc + e * CAND;
    float s[CAND], mx = -3.402823e38f;
    #pragma unroll
    for (int j = 0; j < CAND; ++j) { s[j] = scp[j]; mx = fmaxf(mx, s[j]); }
    float den = 0.f;
    #pragma unroll
    for (int j = 0; j < CAND; ++j) { s[j] = __expf(s[j] - mx); den += s[j]; }
    const float inv = 1.f / den;

    const float* fte = ft + e * (CAND * DD);
    float h[4];
    #pragma unroll
    for (int cc = 0; cc < 4; ++cc) {
        float a = 0.f;
        #pragma unroll
        for (int j = 0; j < CAND; ++j)
            a = fmaf(s[j], fte[j * DD + cc * 32 + lane], a);
        h[cc] = a * inv;
    }
    float d0 = 0.f, d1 = 0.f;
    #pragma unroll
    for (int cc = 0; cc < 4; ++cc) {
        int c = cc * 32 + lane;
        d0 = fmaf(h[cc], Wfc[c * 2 + 0], d0);
        d1 = fmaf(h[cc], Wfc[c * 2 + 1], d1);
    }
    d0 = wave_allreduce(d0);
    d1 = wave_allreduce(d1);
    if (lane == 0) {
        out[e * 2 + 0] = 1.f / (1.f + __expf(-(d0 + bfc[0])));
        out[e * 2 + 1] = 1.f / (1.f + __expf(-(d1 + bfc[1])));
    }
}

// ---------------------------------------------------------------------------
extern "C" void kernel_launch(void* const* d_in, const int* in_sizes, int n_in,
                              void* d_out, int out_size, void* d_ws, size_t ws_size,
                              hipStream_t stream)
{
    const float* feats        = (const float*)d_in[0];
    const int*   edge_members = (const int*)  d_in[1];
    const int*   adj_members  = (const int*)  d_in[2];
    // d_in[3] = ids, d_in[4] = epoch : unused (eval path is unconditional here)
    const float* wq  = (const float*)d_in[5];
    const float* bq  = (const float*)d_in[6];
    const float* wk  = (const float*)d_in[7];
    const float* bk  = (const float*)d_in[8];
    const float* wv  = (const float*)d_in[9];
    const float* bv  = (const float*)d_in[10];
    const float* W1  = (const float*)d_in[11];
    const float* b1  = (const float*)d_in[12];
    const float* W2  = (const float*)d_in[13];
    const float* b2  = (const float*)d_in[14];
    const float* Wfc = (const float*)d_in[15];
    const float* bfc = (const float*)d_in[16];

    const int E = in_sizes[1] / KK;            // 20000
    float* ft = (float*)d_ws;                  // E*5*128 f32
    float* sc = ft + (size_t)E * CAND * DD;    // E*5 f32
    float* out = (float*)d_out;

    const int nrows = E * CAND;                // 100000 (divisible by 16)

    {   // 1) vertex_conv over E + E*R groups (one wave each)
        long ngroups = (long)E * CAND;
        int  blocks  = (int)((ngroups + WAVES_PER_BLOCK - 1) / WAVES_PER_BLOCK);
        vconv_kernel<<<blocks, 32 * WAVES_PER_BLOCK, 0, stream>>>(
            feats, edge_members, adj_members, wq, bq, wk, bk, wv, bv, ft, E);
    }
    {   // 2) EdgeConv scores via WMMA; ~8 tiles per wave to amortize W1 fragments
        edgeconv_kernel<<<98, 32 * WAVES_PER_BLOCK, 0, stream>>>(
            ft, W1, b1, W2, b2, sc, nrows);
    }
    {   // 3) candidate softmax + weighted sum + sigmoid classifier
        int blocks = (E + WAVES_PER_BLOCK - 1) / WAVES_PER_BLOCK;
        final_kernel<<<blocks, 32 * WAVES_PER_BLOCK, 0, stream>>>(
            ft, sc, Wfc, bfc, out, E);
    }
}
